// TransformerLayerShard_2826088480926
// MI455X (gfx1250) — matrix operand order repaired and checked by reading, not compiled, not run on hardware
//
#include <hip/hip_runtime.h>
#include <hip/hip_bf16.h>
#include <math.h>

#define SEQ    2048
#define DMODEL 2048
#define NHEADS 16
#define DPH    128
#define DFF    8192

typedef __attribute__((ext_vector_type(16))) __bf16 v16bf;
typedef __attribute__((ext_vector_type(8)))  float  v8f;

union Frag16 { v16bf v; unsigned int u[8]; };

__device__ __forceinline__ unsigned short f2bf(float f) {
    unsigned int x = __float_as_uint(f);
    unsigned int r = x + 0x7FFFu + ((x >> 16) & 1u);   // round-to-nearest-even
    return (unsigned short)(r >> 16);
}

__device__ __forceinline__ float gelu_tanh(float x) {
    const float k0 = 0.7978845608028654f; // sqrt(2/pi)
    float inner = k0 * (x + 0.044715f * x * x * x);
    return 0.5f * x * (1.0f + tanhf(inner));
}

// ---------------------------------------------------------------------------
// bf16 WMMA GEMM: C(MxN) = A(MxK) * B^T where B is stored N x K (ldb = K-ish
// row stride). Tiles 128x128, BK=32, double-buffered LDS with register
// staging. grid = (N/128, M/128), block = 256 (8 wave32 waves in a 2x4 grid,
// each wave owns a 64x32 sub-tile = 4x2 wmma tiles).
// EPI: 0 -> f32 out, 1 -> bias + gelu -> bf16 out, 2 -> bias -> f32 out.
// All dims assumed multiples of tile sizes (true for every call here).
// ---------------------------------------------------------------------------
template<int EPI>
__global__ __launch_bounds__(256)
void gemm_wmma_bf16(const unsigned short* __restrict__ A, int lda,
                    const unsigned short* __restrict__ B, int ldb,
                    void* __restrict__ C, int ldc, int K,
                    const float* __restrict__ bias)
{
    // Both tiles stored "row-major by fragment owner": As[m][k], Bs[n][k],
    // 128 x 32 bf16 each, double buffered (32 KB LDS total).
    __shared__ unsigned short As[2][128 * 32];
    __shared__ unsigned short Bs[2][128 * 32];

    const int tid  = threadIdx.x;
    const int lane = tid & 31;
    const int wave = tid >> 5;
    const int wm   = (wave & 1) * 64;   // wave M offset within 128 tile
    const int wn   = (wave >> 1) * 32;  // wave N offset within 128 tile
    const int m0   = blockIdx.y * 128;
    const int n0   = blockIdx.x * 128;

    const int r    = lane & 15;
    const int kb8  = (lane >> 4) * 8;   // A-frag K base (16-bit A layout)
    const int kb16 = (lane >> 4) * 16;  // unused; B frags also use kb8 (same K split as A)

    // Per-thread tile-copy coordinates: 512 uint4 (16B) moves per 128x32 tile,
    // 2 per thread; row = idx/4, 16-byte segment = idx%4.
    const int rowC0 = tid >> 2,           segC0 = tid & 3;
    const int rowC1 = (tid + 256) >> 2,   segC1 = (tid + 256) & 3;

    v8f acc[4][2];
#pragma unroll
    for (int i = 0; i < 4; ++i)
#pragma unroll
        for (int j = 0; j < 2; ++j)
#pragma unroll
            for (int e = 0; e < 8; ++e) acc[i][j][e] = 0.0f;

    const int nIter = K >> 5;   // K / 32

    uint4 aR0, aR1, bR0, bR1;

    // ---- prologue: tile 0 -> regs -> LDS[0]
    aR0 = *(const uint4*)&A[(size_t)(m0 + rowC0) * lda + segC0 * 8];
    aR1 = *(const uint4*)&A[(size_t)(m0 + rowC1) * lda + segC1 * 8];
    bR0 = *(const uint4*)&B[(size_t)(n0 + rowC0) * ldb + segC0 * 8];
    bR1 = *(const uint4*)&B[(size_t)(n0 + rowC1) * ldb + segC1 * 8];
    *(uint4*)&As[0][rowC0 * 32 + segC0 * 8] = aR0;
    *(uint4*)&As[0][rowC1 * 32 + segC1 * 8] = aR1;
    *(uint4*)&Bs[0][rowC0 * 32 + segC0 * 8] = bR0;
    *(uint4*)&Bs[0][rowC1 * 32 + segC1 * 8] = bR1;
    __syncthreads();

    for (int it = 0; it < nIter; ++it) {
        const int cur = it & 1;
        const bool hasNext = (it + 1) < nIter;

        // ---- issue next tile's global loads (overlap with WMMA below)
        if (hasNext) {
            const int k0 = (it + 1) << 5;
            aR0 = *(const uint4*)&A[(size_t)(m0 + rowC0) * lda + k0 + segC0 * 8];
            aR1 = *(const uint4*)&A[(size_t)(m0 + rowC1) * lda + k0 + segC1 * 8];
            bR0 = *(const uint4*)&B[(size_t)(n0 + rowC0) * ldb + k0 + segC0 * 8];
            bR1 = *(const uint4*)&B[(size_t)(n0 + rowC1) * ldb + k0 + segC1 * 8];
        }
        // ---- prefetch two steps ahead into L2 (gfx1250 global_prefetch)
        if ((it + 2) < nIter) {
            const int kp = (it + 2) << 5;
            __builtin_prefetch(&A[(size_t)(m0 + rowC0) * lda + kp + segC0 * 8], 0, 1);
            __builtin_prefetch(&B[(size_t)(n0 + rowC0) * ldb + kp + segC0 * 8], 0, 1);
        }

        // ---- B fragments: K {kb8..kb8+7, 16+kb8..16+kb8+7} per lane -> 2x ds_load_b128
        Frag16 bfrag[2];
#pragma unroll
        for (int nt = 0; nt < 2; ++nt) {
            const unsigned short* bp = &Bs[cur][(wn + nt * 16 + r) * 32 + kb8];
            *(uint4*)&bfrag[nt].u[0] = *(const uint4*)(bp);
            *(uint4*)&bfrag[nt].u[4] = *(const uint4*)(bp + 16);
        }

        // ---- A fragments + WMMA (K {kb8..kb8+7, 16+kb8..16+kb8+7})
#pragma unroll
        for (int mt = 0; mt < 4; ++mt) {
            Frag16 afrag;
            const unsigned short* ap = &As[cur][(wm + mt * 16 + r) * 32];
            *(uint4*)&afrag.u[0] = *(const uint4*)(ap + kb8);
            *(uint4*)&afrag.u[4] = *(const uint4*)(ap + 16 + kb8);
#pragma unroll
            for (int nt = 0; nt < 2; ++nt) {
                acc[mt][nt] = __builtin_amdgcn_wmma_f32_16x16x32_bf16(
                    false, afrag.v, false, bfrag[nt].v,
                    (short)0, acc[mt][nt], false, false);
            }
        }

        // ---- commit staged regs into the other LDS buffer
        if (hasNext) {
            const int nxt = cur ^ 1;
            *(uint4*)&As[nxt][rowC0 * 32 + segC0 * 8] = aR0;
            *(uint4*)&As[nxt][rowC1 * 32 + segC1 * 8] = aR1;
            *(uint4*)&Bs[nxt][rowC0 * 32 + segC0 * 8] = bR0;
            *(uint4*)&Bs[nxt][rowC1 * 32 + segC1 * 8] = bR1;
        }
        __syncthreads();
    }

    // ---- epilogue (C/D layout: VGPR rr -> M=rr(+8 for hi lanes), N=lane&15)
    const int mhi = (lane >> 4) * 8;
#pragma unroll
    for (int mt = 0; mt < 4; ++mt)
#pragma unroll
        for (int nt = 0; nt < 2; ++nt)
#pragma unroll
            for (int rr = 0; rr < 8; ++rr) {
                int row = m0 + wm + mt * 16 + rr + mhi;
                int col = n0 + wn + nt * 16 + (lane & 15);
                float v = acc[mt][nt][rr];
                if (EPI == 0) {
                    ((float*)C)[(size_t)row * ldc + col] = v;
                } else if (EPI == 1) {
                    v = gelu_tanh(v + bias[col]);
                    ((unsigned short*)C)[(size_t)row * ldc + col] = f2bf(v);
                } else {
                    ((float*)C)[(size_t)row * ldc + col] = v + bias[col];
                }
            }
}

// ---------------------------------------------------------------------------
// LayerNorm over last dim (2048); writes bf16 output (only bf16 is consumed).
// ---------------------------------------------------------------------------
__global__ __launch_bounds__(256)
void layernorm_kernel(const float* __restrict__ x,
                      const float* __restrict__ scale,
                      const float* __restrict__ offset,
                      unsigned short* __restrict__ xnb)
{
    __shared__ float red[256];
    const int row = blockIdx.x;
    const int tid = threadIdx.x;
    const float* xr = x + (size_t)row * DMODEL;

    float s = 0.0f;
    for (int i = tid; i < DMODEL; i += 256) s += xr[i];
    red[tid] = s; __syncthreads();
    for (int k = 128; k > 0; k >>= 1) { if (tid < k) red[tid] += red[tid + k]; __syncthreads(); }
    float mean = red[0] * (1.0f / DMODEL); __syncthreads();

    float vs = 0.0f;
    for (int i = tid; i < DMODEL; i += 256) { float d = xr[i] - mean; vs += d * d; }
    red[tid] = vs; __syncthreads();
    for (int k = 128; k > 0; k >>= 1) { if (tid < k) red[tid] += red[tid + k]; __syncthreads(); }
    float rstd = rsqrtf(red[0] * (1.0f / DMODEL) + 1e-5f);

    for (int i = tid; i < DMODEL; i += 256) {
        float v = scale[i] * rstd * (xr[i] - mean) + offset[i];
        xnb[(size_t)row * DMODEL + i] = f2bf(v);
    }
}

// f32 [R x C] -> bf16 [C x R] (LDS-tiled transpose, both sides coalesced).
__global__ __launch_bounds__(256)
void transpose_f32_bf16_kernel(const float* __restrict__ in,
                               unsigned short* __restrict__ out, int R, int C)
{
    __shared__ unsigned short tile[32][33];
    const int bx = blockIdx.x * 32;   // input col base
    const int by = blockIdx.y * 32;   // input row base
    const int tx = threadIdx.x, ty = threadIdx.y;   // 32 x 8
#pragma unroll
    for (int j = 0; j < 4; ++j) {
        int y = by + ty + j * 8;
        tile[ty + j * 8][tx] = f2bf(in[(size_t)y * C + bx + tx]);
    }
    __syncthreads();
#pragma unroll
    for (int j = 0; j < 4; ++j) {
        int c = bx + ty + j * 8;      // output row (input col)
        out[(size_t)c * R + by + tx] = tile[tx][ty + j * 8];
    }
}

__global__ __launch_bounds__(256)
void convert_bf16_kernel(const float* __restrict__ in, unsigned short* __restrict__ out, int n)
{
    int i = blockIdx.x * 256 + threadIdx.x;
    if (i < n) out[i] = f2bf(in[i]);
}

// Rotary (GPT-J interleaved per reference) on [SEQ, DMODEL]; emits bf16.
__global__ __launch_bounds__(256)
void rotary_bf16_kernel(const float* __restrict__ in, unsigned short* __restrict__ out)
{
    int idx = blockIdx.x * 256 + threadIdx.x;
    if (idx >= SEQ * DMODEL) return;
    int t = idx / DMODEL;
    int col = idx - t * DMODEL;
    int d = col & (DPH - 1);
    float invf = __powf(10000.0f, -(float)(2 * (d & 63)) / (float)DPH);
    float ang = (float)t * invf;
    float sv = __sinf(ang), cv = __cosf(ang);
    float xv = in[idx];
    float other = (d & 1) ? in[idx - 1] : in[idx + 1];
    float rot = (d & 1) ? other : -other;
    out[idx] = f2bf(rot * sv + xv * cv);
}

// Row softmax over [SEQ] with additive bias; emits bf16 probabilities.
__global__ __launch_bounds__(256)
void softmax_bf16_kernel(const float* __restrict__ logits,
                         const float* __restrict__ bias,
                         unsigned short* __restrict__ probs)
{
    __shared__ float red[256];
    const int t = blockIdx.x;
    const int tid = threadIdx.x;
    const float* L = logits + (size_t)t * SEQ;
    const float* Bb = bias + (size_t)t * SEQ;

    float mx = -3.4e38f;
    for (int i = tid; i < SEQ; i += 256) mx = fmaxf(mx, L[i] + Bb[i]);
    red[tid] = mx; __syncthreads();
    for (int k = 128; k > 0; k >>= 1) { if (tid < k) red[tid] = fmaxf(red[tid], red[tid + k]); __syncthreads(); }
    mx = red[0]; __syncthreads();

    float sum = 0.0f;
    for (int i = tid; i < SEQ; i += 256) sum += __expf(L[i] + Bb[i] - mx);
    red[tid] = sum; __syncthreads();
    for (int k = 128; k > 0; k >>= 1) { if (tid < k) red[tid] += red[tid + k]; __syncthreads(); }
    float inv = 1.0f / red[0];

    for (int i = tid; i < SEQ; i += 256)
        probs[(size_t)t * SEQ + i] = f2bf(__expf(L[i] + Bb[i] - mx) * inv);
}

__global__ __launch_bounds__(256)
void add_kernel(const float* __restrict__ a, const float* __restrict__ b,
                float* __restrict__ out, int n)
{
    int i = blockIdx.x * 256 + threadIdx.x;
    if (i < n) out[i] = a[i] + b[i];
}

// ---------------------------------------------------------------------------
extern "C" void kernel_launch(void* const* d_in, const int* in_sizes, int n_in,
                              void* d_out, int out_size, void* d_ws, size_t ws_size,
                              hipStream_t stream)
{
    (void)in_sizes; (void)n_in; (void)out_size; (void)ws_size;

    const float* x         = (const float*)d_in[0];
    const float* attn_bias = (const float*)d_in[1];
    const float* ln_scale  = (const float*)d_in[2];
    const float* ln_offset = (const float*)d_in[3];
    const float* Wq        = (const float*)d_in[4];
    const float* Wk        = (const float*)d_in[5];
    const float* Wv        = (const float*)d_in[6];
    const float* Wo        = (const float*)d_in[7];
    const float* W1        = (const float*)d_in[8];
    const float* b1        = (const float*)d_in[9];
    const float* W2        = (const float*)d_in[10];
    const float* b2        = (const float*)d_in[11];
    float* out = (float*)d_out;

    // ---- workspace carving
    char* base = (char*)d_ws;
    size_t off = 0;
    auto carve = [&](size_t bytes) -> void* {
        void* p = base + off;
        off = (off + bytes + 255) & ~(size_t)255;
        return p;
    };
    const size_t SD  = (size_t)SEQ * DMODEL;          // 4M elems
    const size_t WFF = (size_t)DMODEL * DFF;          // 16M elems

    unsigned short* xn_b   = (unsigned short*)carve(SD * 2);
    unsigned short* WqT    = (unsigned short*)carve(SD * 2);   // [N][K]
    unsigned short* WkT    = (unsigned short*)carve(SD * 2);
    unsigned short* WvT    = (unsigned short*)carve(SD * 2);
    unsigned short* WoT    = (unsigned short*)carve(SD * 2);
    unsigned short* W1T    = (unsigned short*)carve(WFF * 2);  // [DFF][DMODEL]
    unsigned short* W2T    = (unsigned short*)carve(WFF * 2);  // [DMODEL][DFF]
    float*          Qf     = (float*)carve(SD * 4);
    float*          Kf     = (float*)carve(SD * 4);
    float*          Vf     = (float*)carve(SD * 4);
    unsigned short* Qrb    = (unsigned short*)carve(SD * 2);
    unsigned short* Krb    = (unsigned short*)carve(SD * 2);
    unsigned short* Vt     = (unsigned short*)carve(SD * 2);   // [d][T]
    float*          logits = (float*)carve((size_t)SEQ * SEQ * 4);
    unsigned short* probs  = (unsigned short*)carve((size_t)SEQ * SEQ * 2);
    float*          avec_f = (float*)carve(SD * 4);
    unsigned short* avec_b = (unsigned short*)carve(SD * 2);
    float*          aout_f = (float*)carve(SD * 4);
    unsigned short* h_b    = (unsigned short*)carve((size_t)SEQ * DFF * 2);
    float*          dense  = (float*)carve(SD * 4);

    const int nSD = SEQ * DMODEL;
    dim3 blk(256);
    dim3 tblk(32, 8);

    // 1) LayerNorm -> bf16
    layernorm_kernel<<<SEQ, blk, 0, stream>>>(x, ln_scale, ln_offset, xn_b);

    // 2) weights -> bf16, transposed to [N][K] for the BT GEMM path
    dim3 gT(DMODEL / 32, DMODEL / 32);
    transpose_f32_bf16_kernel<<<gT, tblk, 0, stream>>>(Wq, WqT, DMODEL, DMODEL);
    transpose_f32_bf16_kernel<<<gT, tblk, 0, stream>>>(Wk, WkT, DMODEL, DMODEL);
    transpose_f32_bf16_kernel<<<gT, tblk, 0, stream>>>(Wv, WvT, DMODEL, DMODEL);
    transpose_f32_bf16_kernel<<<gT, tblk, 0, stream>>>(Wo, WoT, DMODEL, DMODEL);
    dim3 gT1(DFF / 32, DMODEL / 32);
    transpose_f32_bf16_kernel<<<gT1, tblk, 0, stream>>>(W1, W1T, DMODEL, DFF);
    dim3 gT2(DMODEL / 32, DFF / 32);
    transpose_f32_bf16_kernel<<<gT2, tblk, 0, stream>>>(W2, W2T, DFF, DMODEL);

    // 3) Q/K/V projections (bf16 WMMA, f32 out)
    dim3 gSq(DMODEL / 128, SEQ / 128);
    gemm_wmma_bf16<0><<<gSq, blk, 0, stream>>>(xn_b, DMODEL, WqT, DMODEL, Qf, DMODEL, DMODEL, nullptr);
    gemm_wmma_bf16<0><<<gSq, blk, 0, stream>>>(xn_b, DMODEL, WkT, DMODEL, Kf, DMODEL, DMODEL, nullptr);
    gemm_wmma_bf16<0><<<gSq, blk, 0, stream>>>(xn_b, DMODEL, WvT, DMODEL, Vf, DMODEL, DMODEL, nullptr);

    // 4) Rotary on Q,K (-> bf16); V -> bf16 transposed to [d][T]
    rotary_bf16_kernel<<<(nSD + 255) / 256, blk, 0, stream>>>(Qf, Qrb);
    rotary_bf16_kernel<<<(nSD + 255) / 256, blk, 0, stream>>>(Kf, Krb);
    transpose_f32_bf16_kernel<<<gT, tblk, 0, stream>>>(Vf, Vt, SEQ, DMODEL);

    // 5) Attention per head (stream-serialized; logits/probs reused)
    dim3 gLog(SEQ / 128, SEQ / 128);   // 2048 x 2048 logits
    dim3 gAV(DPH / 128, SEQ / 128);    // 2048 x 128 attn slice
    for (int h = 0; h < NHEADS; ++h) {
        const unsigned short* Qh = Qrb + h * DPH;                 // [T][d] slice
        const unsigned short* Kh = Krb + h * DPH;                 // B = K, N x K = [T][d]
        const unsigned short* Vh = Vt + (size_t)h * DPH * SEQ;    // B = V^T, N x K = [d][T]
        // logits = Qh (2048x128) * Kh^T -> f32
        gemm_wmma_bf16<0><<<gLog, blk, 0, stream>>>(Qh, DMODEL, Kh, DMODEL, logits, SEQ, DPH, nullptr);
        // softmax(+attn_bias) -> bf16 probs
        softmax_bf16_kernel<<<SEQ, blk, 0, stream>>>(logits, attn_bias, probs);
        // attn_vec slice = probs (2048x2048) * Vh^T(=V) (2048x128)
        gemm_wmma_bf16<0><<<gAV, blk, 0, stream>>>(probs, SEQ, Vh, SEQ, avec_f + h * DPH, DMODEL, SEQ, nullptr);
    }

    // 6) attn_out = attn_vec @ Wo
    convert_bf16_kernel<<<(nSD + 255) / 256, blk, 0, stream>>>(avec_f, avec_b, nSD);
    gemm_wmma_bf16<0><<<gSq, blk, 0, stream>>>(avec_b, DMODEL, WoT, DMODEL, aout_f, DMODEL, DMODEL, nullptr);

    // 7) FFN: h = gelu(xn @ W1 + b1) (bf16 out), dense = h @ W2 + b2 (f32)
    dim3 gF1(DFF / 128, SEQ / 128);
    gemm_wmma_bf16<1><<<gF1, blk, 0, stream>>>(xn_b, DMODEL, W1T, DMODEL, h_b, DFF, DMODEL, b1);
    gemm_wmma_bf16<2><<<gSq, blk, 0, stream>>>(h_b, DFF, W2T, DFF, dense, DMODEL, DFF, b2);

    // 8) residual combine
    add_kernel<<<(nSD + 255) / 256, blk, 0, stream>>>(aout_f, dense, out, nSD);
}
